// FlowEmbeddingMultiHeadCrossAttention_32134945308734
// MI455X (gfx1250) — matrix operand order repaired and checked by reading, not compiled, not run on hardware
//
#include <hip/hip_runtime.h>

// ---------------- types / helpers ----------------
typedef __attribute__((ext_vector_type(16))) __bf16 v16bf;
typedef __attribute__((ext_vector_type(8)))  float  v8f;
typedef __attribute__((ext_vector_type(4)))  unsigned int v4u;

union Frag { v4u u[2]; v16bf v; };

static __device__ __forceinline__ unsigned short f2bf(float f) {
  unsigned int u = __float_as_uint(f);
  u += 0x7FFFu + ((u >> 16) & 1u);      // round-to-nearest-even
  return (unsigned short)(u >> 16);
}

// A-matrix 16x32 bf16 fragment (ISA 7.12.2): lane<16 -> row=l, K {k..k+7, k+16..k+23}
//                                            lane>=16 -> row=l, K {k+8..k+15, k+24..k+31}
static __device__ __forceinline__ v16bf load_fragA(const unsigned short* tile, int stride, int lane) {
  int l = lane & 15, hi = lane >> 4;
  const unsigned short* p = tile + (size_t)l * stride + hi * 8;
  Frag f;
  f.u[0] = *(const v4u*)(p);
  f.u[1] = *(const v4u*)(p + 16);
  return f.v;
}

// B-matrix 32x16 bf16 fragment (same K order as A): lane<16 -> col=l, K {k..k+7, k+16..k+23};
// lane>=16 -> col=l, K {k+8..k+15, k+24..k+31}. tile is [col][k] row-major with given stride.
static __device__ __forceinline__ v16bf load_fragB(const unsigned short* tile, int stride, int lane) {
  int l = lane & 15, hi = lane >> 4;
  const unsigned short* p = tile + (size_t)l * stride + hi * 8;
  Frag f;
  f.u[0] = *(const v4u*)(p);
  f.u[1] = *(const v4u*)(p + 16);
  return f.v;
}

static __device__ __forceinline__ v8f wmma_bf16(v16bf a, v16bf b, v8f c) {
  return __builtin_amdgcn_wmma_f32_16x16x32_bf16(false, a, false, b, (short)0, c, false, false);
}

// ---------------- constants ----------------
#define BB 2
#define NN 4096
#define MM 4096
#define CC 128
#define KS 16
#define ROWS (BB*NN*KS)   // 131072
#define R2   (BB*NN)      // 8192

// ---------------- kernels ----------------
__global__ void zero_kernel(float* p, int n) {
  int i = blockIdx.x * blockDim.x + threadIdx.x;
  if (i < n) p[i] = 0.f;
}

__global__ void knn_kernel(const float* __restrict__ pos1, const float* __restrict__ pos2,
                           int* __restrict__ idx) {
  __shared__ float sp[3 * MM];
  int b = blockIdx.x >> 4;
  int nblk = blockIdx.x & 15;
  const float* p2 = pos2 + (size_t)b * 3 * MM;
  for (int i = threadIdx.x; i < 3 * MM; i += blockDim.x) sp[i] = p2[i];
  __syncthreads();
  int n = nblk * 256 + threadIdx.x;
  const float* p1 = pos1 + (size_t)b * 3 * NN;
  float qx = p1[n], qy = p1[NN + n], qz = p1[2 * NN + n];
  float bd[16]; int bi[16];
#pragma unroll
  for (int j = 0; j < 16; ++j) { bd[j] = 3.4e38f; bi[j] = 0; }
  for (int m = 0; m < MM; ++m) {
    float dx = sp[m] - qx, dy = sp[MM + m] - qy, dz = sp[2 * MM + m] - qz;
    float d = dx * dx + dy * dy + dz * dz;
    if (d < bd[15]) {
      bd[15] = d; bi[15] = m;
#pragma unroll
      for (int j = 15; j > 0; --j) {
        if (bd[j] < bd[j - 1]) {
          float td = bd[j]; bd[j] = bd[j - 1]; bd[j - 1] = td;
          int ti = bi[j]; bi[j] = bi[j - 1]; bi[j - 1] = ti;
        }
      }
    }
  }
  int* o = idx + ((size_t)b * NN + n) * KS;
#pragma unroll
  for (int j = 0; j < 16; ++j) o[j] = bi[j];
}

// pad-convert f32 weight [OUTC x Kin] -> bf16 [OUTC x Kp], zero padded
__global__ void wcvt_kernel(const float* __restrict__ w, unsigned short* __restrict__ out,
                            int OUTC, int Kin, int Kp) {
  int i = blockIdx.x * blockDim.x + threadIdx.x;
  if (i >= OUTC * Kp) return;
  int k = i % Kp, o = i / Kp;
  out[i] = (k < Kin) ? f2bf(w[(size_t)o * Kin + k]) : (unsigned short)0;
}

__global__ void cvt_kernel(const float* __restrict__ in, unsigned short* __restrict__ out,
                           int n, float scale) {
  int i = blockIdx.x * blockDim.x + threadIdx.x;
  if (i < n) out[i] = f2bf(in[i] * scale);
}

// gather [pos_diff(3) | f2g(128) | f1(128) | pad(29)] -> A0 bf16 [ROWS x 288]
__global__ void prep0_kernel(const float* __restrict__ pos1, const float* __restrict__ pos2,
                             const float* __restrict__ f1, const float* __restrict__ f2,
                             const int* __restrict__ idx, unsigned short* __restrict__ A0) {
  size_t i = blockIdx.x * (size_t)blockDim.x + threadIdx.x;
  if (i >= (size_t)ROWS * 288) return;
  int c = (int)(i % 288);
  size_t r = i / 288;
  size_t bn = r / KS;
  int n = (int)(bn % NN);
  int b = (int)(bn / NN);
  float v;
  if (c < 3) {
    int j = idx[r];
    v = pos2[((size_t)b * 3 + c) * MM + j] - pos1[((size_t)b * 3 + c) * NN + n];
  } else if (c < 131) {
    int j = idx[r];
    v = f2[((size_t)b * CC + (c - 3)) * MM + j];
  } else if (c < 259) {
    v = f1[((size_t)b * CC + (c - 131)) * NN + n];
  } else v = 0.f;
  A0[i] = f2bf(v);
}

// C[rows x OUT] = A[rows x Kp](bf16) @ W[OUT x Kp]^T(bf16) (+bias)(+residual); optional raw stats.
// One wave computes a 32x64 tile. Only the streamed A operand is register double-buffered
// (W is L1/L2-resident); B fragments are loop-local so there is no register rotation.
__global__ __launch_bounds__(256, 1)
void gemm_bf16_kernel(const unsigned short* __restrict__ A,
                      const unsigned short* __restrict__ W,
                      const float* __restrict__ bias,
                      const float* __restrict__ residual,
                      float* __restrict__ C,
                      float* __restrict__ ssum, float* __restrict__ ssq,
                      int rows, int Kp, int OUT) {
  int lane = threadIdx.x & 31;
  int wave = blockIdx.x * (blockDim.x >> 5) + (threadIdx.x >> 5);
  int rowTiles = rows >> 5;
  int strips = OUT >> 6;
  if (wave >= rowTiles * strips) return;
  int rt = wave % rowTiles;
  int cs = wave / rowTiles;
  int r0 = rt << 5, c0 = cs << 6;

  v8f acc[2][4];
#pragma unroll
  for (int rr = 0; rr < 2; ++rr)
#pragma unroll
    for (int t = 0; t < 4; ++t)
#pragma unroll
      for (int j = 0; j < 8; ++j) acc[rr][t][j] = 0.f;

  const unsigned short* Arow0 = A + (size_t)r0 * Kp;
  const unsigned short* Arow1 = A + (size_t)(r0 + 16) * Kp;
  const unsigned short* W0 = W + (size_t)(c0 +  0) * Kp;
  const unsigned short* W1 = W + (size_t)(c0 + 16) * Kp;
  const unsigned short* W2 = W + (size_t)(c0 + 32) * Kp;
  const unsigned short* W3 = W + (size_t)(c0 + 48) * Kp;

  v16bf a0 = load_fragA(Arow0, Kp, lane);
  v16bf a1 = load_fragA(Arow1, Kp, lane);

  // all k-steps except the last: preload next A fragments before the WMMAs
  for (int k0 = 0; k0 + 32 < Kp; k0 += 32) {
    __builtin_prefetch(Arow0 + k0 + 64, 0, 1);  // global_prefetch one block ahead
    v16bf na0 = load_fragA(Arow0 + k0 + 32, Kp, lane);
    v16bf na1 = load_fragA(Arow1 + k0 + 32, Kp, lane);
    v16bf b0 = load_fragB(W0 + k0, Kp, lane);
    v16bf b1 = load_fragB(W1 + k0, Kp, lane);
    v16bf b2 = load_fragB(W2 + k0, Kp, lane);
    v16bf b3 = load_fragB(W3 + k0, Kp, lane);

    acc[0][0] = wmma_bf16(a0, b0, acc[0][0]);
    acc[1][0] = wmma_bf16(a1, b0, acc[1][0]);
    acc[0][1] = wmma_bf16(a0, b1, acc[0][1]);
    acc[1][1] = wmma_bf16(a1, b1, acc[1][1]);
    acc[0][2] = wmma_bf16(a0, b2, acc[0][2]);
    acc[1][2] = wmma_bf16(a1, b2, acc[1][2]);
    acc[0][3] = wmma_bf16(a0, b3, acc[0][3]);
    acc[1][3] = wmma_bf16(a1, b3, acc[1][3]);

    a0 = na0; a1 = na1;
  }
  // peeled last k-step
  {
    int k0 = Kp - 32;
    v16bf b0 = load_fragB(W0 + k0, Kp, lane);
    v16bf b1 = load_fragB(W1 + k0, Kp, lane);
    v16bf b2 = load_fragB(W2 + k0, Kp, lane);
    v16bf b3 = load_fragB(W3 + k0, Kp, lane);
    acc[0][0] = wmma_bf16(a0, b0, acc[0][0]);
    acc[1][0] = wmma_bf16(a1, b0, acc[1][0]);
    acc[0][1] = wmma_bf16(a0, b1, acc[0][1]);
    acc[1][1] = wmma_bf16(a1, b1, acc[1][1]);
    acc[0][2] = wmma_bf16(a0, b2, acc[0][2]);
    acc[1][2] = wmma_bf16(a1, b2, acc[1][2]);
    acc[0][3] = wmma_bf16(a0, b3, acc[0][3]);
    acc[1][3] = wmma_bf16(a1, b3, acc[1][3]);
  }

  int l = lane & 15, hi = lane >> 4;
#pragma unroll
  for (int t = 0; t < 4; ++t) {
    int col = c0 + t * 16 + l;
    float bv = bias ? bias[col] : 0.f;
    float s = 0.f, q = 0.f;
#pragma unroll
    for (int rr = 0; rr < 2; ++rr) {
#pragma unroll
      for (int j = 0; j < 8; ++j) {
        int row = r0 + rr * 16 + j + hi * 8;
        float v = acc[rr][t][j];
        s += v; q += v * v;
        v += bv;
        if (residual) v += residual[(size_t)row * OUT + col];
        C[(size_t)row * OUT + col] = v;
      }
    }
    if (ssum) {
      s += __shfl_down(s, 16);
      q += __shfl_down(q, 16);
      if (hi == 0) { atomicAdd(&ssum[col], s); atomicAdd(&ssq[col], q); }
    }
  }
}

__global__ void bnfin_kernel(const float* ssum, const float* ssq, float* mu, float* rs,
                             int Cn, float inv_cnt) {
  int c = blockIdx.x * blockDim.x + threadIdx.x;
  if (c >= Cn) return;
  float m = ssum[c] * inv_cnt;
  float var = ssq[c] * inv_cnt - m * m;
  mu[c] = m;
  rs[c] = rsqrtf(var + 1e-5f);
}

__global__ void prepbn_kernel(const float* __restrict__ y, const float* __restrict__ mu,
                              const float* __restrict__ rs, const float* __restrict__ g,
                              const float* __restrict__ bta, unsigned short* __restrict__ out,
                              size_t total, int Cn) {
  size_t i = blockIdx.x * (size_t)blockDim.x + threadIdx.x;
  if (i >= total) return;
  int c = (int)(i % Cn);
  float v = (y[i] - mu[c]) * rs[c] * g[c] + bta[c];
  out[i] = f2bf(fmaxf(v, 0.f));
}

// BN+ReLU on y2 then max/mean pool over k -> feat[R2 x 512] (f32 + bf16)
__global__ void pool_kernel(const float* __restrict__ y2, const float* __restrict__ mu,
                            const float* __restrict__ rs, const float* __restrict__ g,
                            const float* __restrict__ bta,
                            float* __restrict__ feat, unsigned short* __restrict__ featb) {
  int i = blockIdx.x * blockDim.x + threadIdx.x;
  if (i >= R2 * 256) return;
  int c = i % 256;
  size_t row = i / 256;
  float mx = -3.4e38f, sm = 0.f;
  for (int kk = 0; kk < KS; ++kk) {
    float v = y2[((size_t)row * KS + kk) * 256 + c];
    v = fmaxf((v - mu[c]) * rs[c] * g[c] + bta[c], 0.f);
    mx = fmaxf(mx, v); sm += v;
  }
  float mn = sm * (1.f / 16.f);
  feat[row * 512 + c] = mx;
  feat[row * 512 + 256 + c] = mn;
  featb[row * 512 + c] = f2bf(mx);
  featb[row * 512 + 256 + c] = f2bf(mn);
}

// feature2 (B,C,M) -> bf16 rows [(b*M+m) x C]
__global__ void f2prep_kernel(const float* __restrict__ f2, unsigned short* __restrict__ out) {
  int i = blockIdx.x * blockDim.x + threadIdx.x;
  if (i >= R2 * CC) return;
  int c = i % CC;
  size_t row = i / CC;
  int m = (int)(row % MM);
  int b = (int)(row / MM);
  out[i] = f2bf(f2[((size_t)b * CC + c) * MM + m]);
}

// flash attention: one wave per (b, h, 16-query tile); Q pre-scaled by 1/sqrt(64).
// K fragments register double-buffered across the key loop.
__global__ __launch_bounds__(32, 1)
void attn_kernel(const unsigned short* __restrict__ Qb,
                 const unsigned short* __restrict__ Kb,
                 const unsigned short* __restrict__ Vb,
                 float* __restrict__ Of) {
  __shared__ __align__(16) unsigned short ldsP[16 * 32];
  __shared__ __align__(16) unsigned short ldsVT[64 * 32];
  int lane = threadIdx.x;
  int blk = blockIdx.x;
  int qt = blk & 255;
  int h = (blk >> 8) & 3;
  int b = blk >> 10;
  int q0 = qt << 4;
  int l = lane & 15, hi = lane >> 4;

  const unsigned short* Qh = Qb + ((size_t)b * NN + q0) * 256 + h * 64;
  v16bf qa0 = load_fragA(Qh, 256, lane);
  v16bf qa1 = load_fragA(Qh + 32, 256, lane);

  v8f o0, o1, o2, o3;
  float rowm[8], rowsum[8];
#pragma unroll
  for (int j = 0; j < 8; ++j) {
    o0[j] = o1[j] = o2[j] = o3[j] = 0.f;
    rowm[j] = -1e30f; rowsum[j] = 0.f;
  }

  const unsigned short* KhBase = Kb + ((size_t)b * MM) * 256 + h * 64;
  v16bf kb00 = load_fragB(KhBase, 256, lane);
  v16bf kb01 = load_fragB(KhBase + 32, 256, lane);
  v16bf kb10 = load_fragB(KhBase + 16 * 256, 256, lane);
  v16bf kb11 = load_fragB(KhBase + 16 * 256 + 32, 256, lane);

  for (int m0 = 0; m0 < MM; m0 += 32) {
    int mn = (m0 + 32 < MM) ? (m0 + 32) : m0;
    const unsigned short* Khn = KhBase + (size_t)mn * 256;
    v16bf n00 = load_fragB(Khn, 256, lane);
    v16bf n01 = load_fragB(Khn + 32, 256, lane);
    v16bf n10 = load_fragB(Khn + 16 * 256, 256, lane);
    v16bf n11 = load_fragB(Khn + 16 * 256 + 32, 256, lane);
    __builtin_prefetch(Vb + ((size_t)b * MM + mn + lane) * 256 + h * 64, 0, 1);

    v8f s0, s1;
#pragma unroll
    for (int j = 0; j < 8; ++j) { s0[j] = 0.f; s1[j] = 0.f; }
    s0 = wmma_bf16(qa0, kb00, s0); s0 = wmma_bf16(qa1, kb01, s0);
    s1 = wmma_bf16(qa0, kb10, s1); s1 = wmma_bf16(qa1, kb11, s1);
    kb00 = n00; kb01 = n01; kb10 = n10; kb11 = n11;

    __syncthreads();  // protect LDS reuse (single wave -> barrier is S_NOP + waits)
    // stage V tile transposed: ldsVT[d][key]
    const unsigned short* Vh = Vb + ((size_t)b * MM + m0 + lane) * 256 + h * 64;
#pragma unroll
    for (int d = 0; d < 64; ++d) ldsVT[d * 32 + lane] = Vh[d];

    // online softmax (rows split: vgpr j -> row j (lanes 0-15), row j+8 (lanes 16-31))
#pragma unroll
    for (int j = 0; j < 8; ++j) {
      float mx = fmaxf(s0[j], s1[j]);
      mx = fmaxf(mx, __shfl_xor(mx, 1));
      mx = fmaxf(mx, __shfl_xor(mx, 2));
      mx = fmaxf(mx, __shfl_xor(mx, 4));
      mx = fmaxf(mx, __shfl_xor(mx, 8));
      float mnew = fmaxf(rowm[j], mx);
      float corr = __expf(rowm[j] - mnew);
      rowm[j] = mnew;
      float p0 = __expf(s0[j] - mnew);
      float p1 = __expf(s1[j] - mnew);
      float rsm = p0 + p1;
      rsm += __shfl_xor(rsm, 1);
      rsm += __shfl_xor(rsm, 2);
      rsm += __shfl_xor(rsm, 4);
      rsm += __shfl_xor(rsm, 8);
      rowsum[j] = rowsum[j] * corr + rsm;
      int row = j + hi * 8;
      ldsP[row * 32 + l] = f2bf(p0);
      ldsP[row * 32 + 16 + l] = f2bf(p1);
      o0[j] *= corr; o1[j] *= corr; o2[j] *= corr; o3[j] *= corr;
    }
    __syncthreads();

    v16bf pa = load_fragA(ldsP, 32, lane);
    o0 = wmma_bf16(pa, load_fragB(ldsVT + 0 * 16 * 32, 32, lane), o0);
    o1 = wmma_bf16(pa, load_fragB(ldsVT + 1 * 16 * 32, 32, lane), o1);
    o2 = wmma_bf16(pa, load_fragB(ldsVT + 2 * 16 * 32, 32, lane), o2);
    o3 = wmma_bf16(pa, load_fragB(ldsVT + 3 * 16 * 32, 32, lane), o3);
  }

#pragma unroll
  for (int j = 0; j < 8; ++j) {
    int row = q0 + j + hi * 8;
    float inv = 1.f / rowsum[j];
    float* op = Of + ((size_t)b * NN + row) * 256 + h * 64 + l;
    op[0]  = o0[j] * inv;
    op[16] = o1[j] * inv;
    op[32] = o2[j] * inv;
    op[48] = o3[j] * inv;
  }
}

// LayerNorm over 512 dims -> bf16
__global__ void ln_kernel(const float* __restrict__ y, const float* __restrict__ g,
                          const float* __restrict__ bta, unsigned short* __restrict__ out) {
  __shared__ float red[256];
  int row = blockIdx.x, t = threadIdx.x;
  const float* yr = y + (size_t)row * 512;
  float v0 = yr[t], v1 = yr[t + 256];
  red[t] = v0 + v1;
  __syncthreads();
  for (int off = 128; off > 0; off >>= 1) {
    if (t < off) red[t] += red[t + off];
    __syncthreads();
  }
  float mu = red[0] * (1.f / 512.f);
  __syncthreads();
  float d0 = v0 - mu, d1 = v1 - mu;
  red[t] = d0 * d0 + d1 * d1;
  __syncthreads();
  for (int off = 128; off > 0; off >>= 1) {
    if (t < off) red[t] += red[t + off];
    __syncthreads();
  }
  float rsig = rsqrtf(red[0] * (1.f / 512.f) + 1e-5f);
  out[(size_t)row * 512 + t]       = f2bf(d0 * rsig * g[t] + bta[t]);
  out[(size_t)row * 512 + t + 256] = f2bf(d1 * rsig * g[t + 256] + bta[t + 256]);
}

// d_out = [pos1 flat (B,3,N)] ++ [y^T (B,256,N)]
__global__ void finout_kernel(const float* __restrict__ pos1, const float* __restrict__ fin,
                              float* __restrict__ out) {
  size_t i = blockIdx.x * (size_t)blockDim.x + threadIdx.x;
  if (i < (size_t)BB * 3 * NN) { out[i] = pos1[i]; return; }
  size_t j = i - (size_t)BB * 3 * NN;
  if (j >= (size_t)BB * 256 * NN) return;
  int n = (int)(j % NN);
  int c = (int)((j / NN) % 256);
  int b = (int)(j / ((size_t)NN * 256));
  out[i] = fin[(((size_t)b * NN) + n) * 256 + c];
}

// ---------------- launch ----------------
static inline unsigned gdiv(size_t n, unsigned b) { return (unsigned)((n + b - 1) / b); }
static inline unsigned gemm_blocks(int rows, int OUT) {
  size_t waves = (size_t)(rows / 32) * (OUT / 64);
  return (unsigned)((waves + 7) / 8);
}

extern "C" void kernel_launch(void* const* d_in, const int* in_sizes, int n_in,
                              void* d_out, int out_size, void* d_ws, size_t ws_size,
                              hipStream_t stream) {
  const float* pos1 = (const float*)d_in[0];
  const float* pos2 = (const float*)d_in[1];
  const float* f1   = (const float*)d_in[2];
  const float* f2   = (const float*)d_in[3];
  const float* w0   = (const float*)d_in[4];
  const float* g0   = (const float*)d_in[5];
  const float* b0   = (const float*)d_in[6];
  const float* w1   = (const float*)d_in[7];
  const float* g1   = (const float*)d_in[8];
  const float* b1   = (const float*)d_in[9];
  const float* w2   = (const float*)d_in[10];
  const float* g2   = (const float*)d_in[11];
  const float* b2   = (const float*)d_in[12];
  const float* qw   = (const float*)d_in[13];
  const float* qb   = (const float*)d_in[14];
  const float* kw   = (const float*)d_in[15];
  const float* kb   = (const float*)d_in[16];
  const float* vw   = (const float*)d_in[17];
  const float* vb   = (const float*)d_in[18];
  const float* ow   = (const float*)d_in[19];
  const float* ob   = (const float*)d_in[20];
  const float* lng  = (const float*)d_in[21];
  const float* lnb  = (const float*)d_in[22];
  const float* fw   = (const float*)d_in[23];
  const float* fb   = (const float*)d_in[24];

  char* ws = (char*)d_ws;
  size_t cur = 0;
  auto take = [&](size_t bytes) -> char* {
    char* p = ws + cur;
    cur += (bytes + 255) & ~(size_t)255;
    return p;
  };

  float* stats = (float*)take(2048 * 4);
  float* sum0 = stats,        *sq0 = stats + 128, *mu0 = stats + 256, *rs0 = stats + 384;
  float* sum1 = stats + 512,  *sq1 = stats + 640, *mu1 = stats + 768, *rs1 = stats + 896;
  float* sum2 = stats + 1024, *sq2 = stats + 1280, *mu2 = stats + 1536, *rs2 = stats + 1792;

  int* idx = (int*)take((size_t)ROWS * 4);

  unsigned short* w0b = (unsigned short*)take((size_t)128 * 288 * 2);
  unsigned short* w1b = (unsigned short*)take((size_t)128 * 128 * 2);
  unsigned short* w2b = (unsigned short*)take((size_t)256 * 128 * 2);
  unsigned short* qwb = (unsigned short*)take((size_t)256 * 512 * 2);
  unsigned short* kwb = (unsigned short*)take((size_t)256 * 128 * 2);
  unsigned short* vwb = (unsigned short*)take((size_t)256 * 128 * 2);
  unsigned short* owb = (unsigned short*)take((size_t)512 * 256 * 2);
  unsigned short* fwb = (unsigned short*)take((size_t)256 * 512 * 2);

  char* reg1 = take((size_t)ROWS * 288 * 2);   // 75.5 MB: A0 / y1 / QKV
  char* reg2 = take((size_t)ROWS * 128 * 4);   // 67.1 MB: y0 / A2 / O,LN,final
  char* reg3 = take((size_t)ROWS * 128 * 2);   // 33.6 MB: A1 / feat+featb
  char* reg4 = take((size_t)ROWS * 256 * 4);   // 134.2 MB: y2
  unsigned short* f2b = (unsigned short*)take((size_t)R2 * 128 * 2);

  unsigned short* A0 = (unsigned short*)reg1;
  float* y0 = (float*)reg2;
  unsigned short* A1 = (unsigned short*)reg3;
  float* y1 = (float*)reg1;                           // after A0 dead
  unsigned short* A2 = (unsigned short*)reg2;         // after y0 dead
  float* y2 = (float*)reg4;
  float* feat = (float*)reg3;                         // after A1 dead
  unsigned short* featb = (unsigned short*)(reg3 + 16777216);
  float* Qf = (float*)reg1;                           // after y1 dead
  float* Kf = (float*)(reg1 + 8388608);
  float* Vf = (float*)(reg1 + 16777216);
  unsigned short* Qbq = (unsigned short*)(reg1 + 25165824);
  unsigned short* Kbq = (unsigned short*)(reg1 + 29360128);
  unsigned short* Vbq = (unsigned short*)(reg1 + 33554432);
  float* Of = (float*)reg2;                           // after A2 dead
  unsigned short* Ob = (unsigned short*)(reg2 + 8388608);
  float* yo = (float*)(reg2 + 12582912);
  unsigned short* ylnb = (unsigned short*)(reg2 + 29360128);
  float* ffin = (float*)(reg2 + 37748736);

  // 1. zero stats
  zero_kernel<<<8, 256, 0, stream>>>(stats, 2048);
  // 2. KNN
  knn_kernel<<<32, 256, 0, stream>>>(pos1, pos2, idx);
  // 3. weight conversion (qw NOT pre-scaled; Q scaled at f32->bf16 step)
  wcvt_kernel<<<gdiv(128 * 288, 256), 256, 0, stream>>>(w0, w0b, 128, 259, 288);
  wcvt_kernel<<<gdiv(128 * 128, 256), 256, 0, stream>>>(w1, w1b, 128, 128, 128);
  wcvt_kernel<<<gdiv(256 * 128, 256), 256, 0, stream>>>(w2, w2b, 256, 128, 128);
  wcvt_kernel<<<gdiv(256 * 512, 256), 256, 0, stream>>>(qw, qwb, 256, 512, 512);
  wcvt_kernel<<<gdiv(256 * 128, 256), 256, 0, stream>>>(kw, kwb, 256, 128, 128);
  wcvt_kernel<<<gdiv(256 * 128, 256), 256, 0, stream>>>(vw, vwb, 256, 128, 128);
  wcvt_kernel<<<gdiv(512 * 256, 256), 256, 0, stream>>>(ow, owb, 512, 256, 256);
  wcvt_kernel<<<gdiv(256 * 512, 256), 256, 0, stream>>>(fw, fwb, 256, 512, 512);
  // 4. gather -> A0
  prep0_kernel<<<gdiv((size_t)ROWS * 288, 256), 256, 0, stream>>>(pos1, pos2, f1, f2, idx, A0);
  // 5. MLP layer 0
  gemm_bf16_kernel<<<gemm_blocks(ROWS, 128), 256, 0, stream>>>(
      A0, w0b, nullptr, nullptr, y0, sum0, sq0, ROWS, 288, 128);
  bnfin_kernel<<<1, 128, 0, stream>>>(sum0, sq0, mu0, rs0, 128, 1.f / ROWS);
  prepbn_kernel<<<gdiv((size_t)ROWS * 128, 256), 256, 0, stream>>>(
      y0, mu0, rs0, g0, b0, A1, (size_t)ROWS * 128, 128);
  // 6. MLP layer 1
  gemm_bf16_kernel<<<gemm_blocks(ROWS, 128), 256, 0, stream>>>(
      A1, w1b, nullptr, nullptr, y1, sum1, sq1, ROWS, 128, 128);
  bnfin_kernel<<<1, 128, 0, stream>>>(sum1, sq1, mu1, rs1, 128, 1.f / ROWS);
  prepbn_kernel<<<gdiv((size_t)ROWS * 128, 256), 256, 0, stream>>>(
      y1, mu1, rs1, g1, b1, A2, (size_t)ROWS * 128, 128);
  // 7. MLP layer 2
  gemm_bf16_kernel<<<gemm_blocks(ROWS, 256), 256, 0, stream>>>(
      A2, w2b, nullptr, nullptr, y2, sum2, sq2, ROWS, 128, 256);
  bnfin_kernel<<<1, 256, 0, stream>>>(sum2, sq2, mu2, rs2, 256, 1.f / ROWS);
  // 8. BN+ReLU + max/mean pool -> feat
  pool_kernel<<<gdiv((size_t)R2 * 256, 256), 256, 0, stream>>>(y2, mu2, rs2, g2, b2, feat, featb);
  // 9. Q/K/V projections
  f2prep_kernel<<<gdiv((size_t)R2 * 128, 256), 256, 0, stream>>>(f2, f2b);
  gemm_bf16_kernel<<<gemm_blocks(R2, 256), 256, 0, stream>>>(
      featb, qwb, qb, nullptr, Qf, nullptr, nullptr, R2, 512, 256);
  cvt_kernel<<<gdiv((size_t)R2 * 256, 256), 256, 0, stream>>>(Qf, Qbq, R2 * 256, 0.125f); // 1/sqrt(64)
  gemm_bf16_kernel<<<gemm_blocks(R2, 256), 256, 0, stream>>>(
      f2b, kwb, kb, nullptr, Kf, nullptr, nullptr, R2, 128, 256);
  cvt_kernel<<<gdiv((size_t)R2 * 256, 256), 256, 0, stream>>>(Kf, Kbq, R2 * 256, 1.f);
  gemm_bf16_kernel<<<gemm_blocks(R2, 256), 256, 0, stream>>>(
      f2b, vwb, vb, nullptr, Vf, nullptr, nullptr, R2, 128, 256);
  cvt_kernel<<<gdiv((size_t)R2 * 256, 256), 256, 0, stream>>>(Vf, Vbq, R2 * 256, 1.f);
  // 10. flash attention
  attn_kernel<<<2048, 32, 0, stream>>>(Qbq, Kbq, Vbq, Of);
  cvt_kernel<<<gdiv((size_t)R2 * 256, 256), 256, 0, stream>>>(Of, Ob, R2 * 256, 1.f);
  // 11. output projection + residual
  gemm_bf16_kernel<<<gemm_blocks(R2, 512), 256, 0, stream>>>(
      Ob, owb, ob, feat, yo, nullptr, nullptr, R2, 256, 512);
  // 12. LayerNorm
  ln_kernel<<<R2, 256, 0, stream>>>(yo, lng, lnb, ylnb);
  // 13. final linear
  gemm_bf16_kernel<<<gemm_blocks(R2, 256), 256, 0, stream>>>(
      ylnb, fwb, fb, nullptr, ffin, nullptr, nullptr, R2, 512, 256);
  // 14. pack output (pos1 passthrough + transposed features)
  finout_kernel<<<gdiv((size_t)BB * 3 * NN + (size_t)BB * 256 * NN, 256), 256, 0, stream>>>(
      pos1, ffin, (float*)d_out);
}
